// My_Net_9457517986565
// MI455X (gfx1250) — matrix operand order repaired and checked by reading, not compiled, not run on hardware
//
#include <hip/hip_runtime.h>

#define H 128
#define BN_EPS 1e-5f

#define N_DR 8000
#define N_P 20000
#define N_DIS 5000
#define N_MF 3000
#define N_BP 8000
#define N_CC 2000
#define N_PATH 2392
#define NB 65536

#define FLAG_RELU 1
#define FLAG_ACCUM 2
#define FLAG_RESID 4

typedef __attribute__((ext_vector_type(16))) __bf16 v16bf;
typedef __attribute__((ext_vector_type(8)))  float v8f;

union FragBF { v16bf v; uint4 q[2]; };

__device__ __forceinline__ unsigned short f2bf(float f) {
    unsigned int u = __float_as_uint(f);
    unsigned int r = (u + 0x7FFFu + ((u >> 16) & 1u)) >> 16;
    return (unsigned short)r;
}
__device__ __forceinline__ unsigned int pack2(float a, float b) {
    return (unsigned int)f2bf(a) | ((unsigned int)f2bf(b) << 16);
}

// ---------------------------------------------------------------------------
// Generic dual-source WMMA GEMM:
//   out[M,N] (+)= epilogue( A1[rows1][:,0:K1] @ W1 + A2[rows2][:,0:K2] @ W2 + bias )
// Block = 256 threads (8 wave32), 64x64 tile; each wave computes 16x32.
// Fast (uniform-branch) vectorized staging for interior tiles.
// ---------------------------------------------------------------------------
__global__ __launch_bounds__(256)
void wmma_gemm_kernel(const float* __restrict__ A1, const int* __restrict__ rows1,
                      int lda1, int K1, const float* __restrict__ W1,
                      const float* __restrict__ A2, const int* __restrict__ rows2,
                      int lda2, int K2, const float* __restrict__ W2,
                      const float* __restrict__ bias, const float* __restrict__ resid,
                      float* __restrict__ out, int M, int N, int flags)
{
    __shared__ unsigned short As[64][40];   // padded rows: 80B stride, 16B aligned
    __shared__ unsigned short Bs[64][40];   // W tile stored transposed: Bs[n][k]

    const int tid  = threadIdx.x;
    const int lane = tid & 31;
    const int wave = tid >> 5;
    const int wm   = wave & 3;      // 0..3 -> row offset 16*wm
    const int wn   = wave >> 2;     // 0..1 -> col offset 32*wn
    const int half = lane >> 4;
    const int l16  = lane & 15;
    const int bm = blockIdx.y * 64;
    const int bn = blockIdx.x * 64;

    const bool mfull = (bm + 64 <= M);
    const bool nfull = (bn + 64 <= N);

    v8f c0 = {0.f,0.f,0.f,0.f,0.f,0.f,0.f,0.f};
    v8f c1 = {0.f,0.f,0.f,0.f,0.f,0.f,0.f,0.f};

    const int ar_l = tid >> 2;          // 0..63 : A-tile row this thread loads
    const int ac_l = (tid & 3) * 8;     // 0,8,16,24
    const int n_l  = tid & 63;          // W-tile column this thread loads
    const int kc_l = (tid >> 6) * 8;    // 0,8,16,24

    for (int part = 0; part < 2; ++part) {
        const float* A = part ? A2 : A1;
        if (A == nullptr) continue;
        const int* rows = part ? rows2 : rows1;
        const int lda   = part ? lda2 : lda1;
        const int K     = part ? K2 : K1;
        const float* W  = part ? W2 : W1;

        const float* arow = nullptr;
        {
            int gr = bm + ar_l;
            if (gr < M) {
                long rr = rows ? (long)rows[gr] : (long)gr;
                arow = A + rr * (long)lda;
            }
        }

        for (int k0 = 0; k0 < K; k0 += 32) {
            const bool kfull = (k0 + 32 <= K);
            // ---- stage A tile (64x32 f32 -> bf16) ----
            if (mfull && kfull) {
                const float* ap = arow + k0 + ac_l;
                float4 v0 = *(const float4*)ap;
                float4 v1 = *(const float4*)(ap + 4);
                uint4 q;
                q.x = pack2(v0.x, v0.y); q.y = pack2(v0.z, v0.w);
                q.z = pack2(v1.x, v1.y); q.w = pack2(v1.z, v1.w);
                *(uint4*)&As[ar_l][ac_l] = q;
            } else {
                #pragma unroll
                for (int j = 0; j < 8; ++j) {
                    int k = k0 + ac_l + j;
                    float v = (arow != nullptr && k < K) ? arow[k] : 0.f;
                    As[ar_l][ac_l + j] = f2bf(v);
                }
            }
            // ---- stage W tile transposed: Bs[n][k] = W[k0+k][bn+n] ----
            if (kfull && nfull) {
                const float* wp = W + (long)(k0 + kc_l) * N + (bn + n_l);
                float w[8];
                #pragma unroll
                for (int j = 0; j < 8; ++j) w[j] = wp[(long)j * N];
                uint4 q;
                q.x = pack2(w[0], w[1]); q.y = pack2(w[2], w[3]);
                q.z = pack2(w[4], w[5]); q.w = pack2(w[6], w[7]);
                *(uint4*)&Bs[n_l][kc_l] = q;
            } else {
                #pragma unroll
                for (int j = 0; j < 8; ++j) {
                    int k = kc_l + j;
                    float v = 0.f;
                    if ((k0 + k) < K && (bn + n_l) < N) v = W[(long)(k0 + k) * N + (bn + n_l)];
                    Bs[n_l][k] = f2bf(v);
                }
            }
            __syncthreads();

            FragBF a, b0, b1;
            const unsigned short* ap = &As[wm * 16 + l16][0];
            a.q[0] = *(const uint4*)(ap + half * 16);
            a.q[1] = *(const uint4*)(ap + half * 16 + 8);
            const unsigned short* bp0 = &Bs[wn * 32 + l16][0];
            b0.q[0] = *(const uint4*)(bp0 + half * 16);
            b0.q[1] = *(const uint4*)(bp0 + half * 16 + 8);
            const unsigned short* bp1 = &Bs[wn * 32 + 16 + l16][0];
            b1.q[0] = *(const uint4*)(bp1 + half * 16);
            b1.q[1] = *(const uint4*)(bp1 + half * 16 + 8);

            c0 = __builtin_amdgcn_wmma_f32_16x16x32_bf16(false, a.v, false, b0.v,
                                                         (short)0, c0, false, false);
            c1 = __builtin_amdgcn_wmma_f32_16x16x32_bf16(false, a.v, false, b1.v,
                                                         (short)0, c1, false, false);
            __syncthreads();
        }
    }

    const int row0 = bm + wm * 16 + half * 8;
    const int col0 = bn + wn * 32 + l16;
    #pragma unroll
    for (int r = 0; r < 8; ++r) {
        int m = row0 + r;
        if (m >= M) continue;
        #pragma unroll
        for (int t = 0; t < 2; ++t) {
            int col = col0 + t * 16;
            if (col >= N) continue;
            float v = t ? c1[r] : c0[r];
            if (bias) v += bias[col];
            if (flags & FLAG_RELU) v = fmaxf(v, 0.f);
            if (flags & FLAG_RESID) v += resid[(long)m * N + col];
            long oi = (long)m * N + col;
            if (flags & FLAG_ACCUM) out[oi] += v; else out[oi] = v;
        }
    }
}

// ---------------------------------------------------------------------------
// Sparse / elementwise helpers
// ---------------------------------------------------------------------------
__global__ void fill_kernel(float* p, float v, long n) {
    long i = (long)blockIdx.x * blockDim.x + threadIdx.x;
    if (i < n) p[i] = v;
}

__global__ void scatter_add_kernel(const float* __restrict__ feat, const int* __restrict__ src,
                                   const int* __restrict__ dst, float* __restrict__ agg, long tot) {
    long i = (long)blockIdx.x * blockDim.x + threadIdx.x;
    if (i >= tot) return;
    long e = i >> 7; int f = (int)(i & (H - 1));
    atomicAdd(&agg[(long)dst[e] * H + f], feat[(long)src[e] * H + f]);
}

// feat >= 0 (post-ReLU) so uint-bit-pattern max over a zero-initialized buffer
// is exact segment_max with DGL's zero-in-degree -> 0 convention.
__global__ void scatter_max_kernel(const float* __restrict__ feat, const int* __restrict__ src,
                                   const int* __restrict__ dst, float* __restrict__ agg, long tot) {
    long i = (long)blockIdx.x * blockDim.x + threadIdx.x;
    if (i >= tot) return;
    long e = i >> 7; int f = (int)(i & (H - 1));
    float v = feat[(long)src[e] * H + f];
    atomicMax((unsigned int*)&agg[(long)dst[e] * H + f], __float_as_uint(v));
}

__global__ void bias_relu_kernel(const float* __restrict__ W, const float* __restrict__ b,
                                 float* __restrict__ o, long n) {
    long i = (long)blockIdx.x * blockDim.x + threadIdx.x;
    if (i < n) o[i] = fmaxf(W[i] + b[i & (H - 1)], 0.f);
}

__global__ void concat_kernel(const float* a, const float* b, const float* c, const float* d,
                              float* o, int M, int nparts) {
    long i = (long)blockIdx.x * blockDim.x + threadIdx.x;
    int ldo = nparts * H;
    long tot = (long)M * ldo;
    if (i >= tot) return;
    long m = i / ldo; int col = (int)(i % ldo);
    int part = col / H, f = col % H;
    const float* s = (part == 0) ? a : (part == 1) ? b : (part == 2) ? c : d;
    o[i] = s[m * H + f];
}

__global__ void bn_stats_kernel(const float* __restrict__ x, float* __restrict__ sums,
                                float* __restrict__ sqs, int rows_per_block, int C) {
    int c = threadIdx.x;
    if (c >= C) return;
    long rbase = (long)blockIdx.x * rows_per_block;
    float s = 0.f, q = 0.f;
    for (int r = 0; r < rows_per_block; ++r) {
        float v = x[(rbase + r) * C + c];
        s += v; q += v * v;
    }
    atomicAdd(&sums[c], s);
    atomicAdd(&sqs[c], q);
}

__global__ void bn_apply_kernel(float* __restrict__ x, const float* __restrict__ sums,
                                const float* __restrict__ sqs, const float* __restrict__ g,
                                const float* __restrict__ be, long n, int C, float invB) {
    long i = (long)blockIdx.x * blockDim.x + threadIdx.x;
    if (i >= n) return;
    int c = (int)(i % C);
    float m = sums[c] * invB;
    float var = sqs[c] * invB - m * m;
    float y = g[c] * (x[i] - m) * rsqrtf(var + BN_EPS) + be[c];
    x[i] = fmaxf(y, 0.f);
}

__global__ void out_kernel(const float* __restrict__ h3, const float* __restrict__ Wo,
                           const float* __restrict__ bo, float* __restrict__ out, int Brows) {
    int i = blockIdx.x * blockDim.x + threadIdx.x;
    if (i >= Brows) return;
    float s = bo[0];
    #pragma unroll
    for (int j = 0; j < 64; ++j) s += h3[(long)i * 64 + j] * Wo[j];
    out[i] = 1.f / (1.f + __expf(-s));
}

// ---------------------------------------------------------------------------
extern "C" void kernel_launch(void* const* d_in, const int* in_sizes, int n_in,
                              void* d_out, int out_size, void* d_ws, size_t ws_size,
                              hipStream_t stream) {
    (void)n_in; (void)out_size; (void)ws_size;
    const int*   x_dr    = (const int*)d_in[0];
    const int*   x_p     = (const int*)d_in[1];
    const float* finger  = (const float*)d_in[2];
    const float* seqf    = (const float*)d_in[3];
    const float* disease = (const float*)d_in[4];
    // d_in[5..8]: identity features — algebraically eliminated.
    const int* e_dtdr = (const int*)d_in[9];   int E_dtdr = in_sizes[9]  / 2;
    const int* e_dmdr = (const int*)d_in[10];  int E_dmdr = in_sizes[10] / 2;
    const int* e_dp   = (const int*)d_in[11];  int E_dp   = in_sizes[11] / 2;
    const int* e_drtd = (const int*)d_in[12];  int E_drtd = in_sizes[12] / 2;
    const int* e_drmd = (const int*)d_in[13];  int E_drmd = in_sizes[13] / 2;
    const int* e_pd   = (const int*)d_in[14];  int E_pd   = in_sizes[14] / 2;
    const int* e_ddi  = (const int*)d_in[15];  int E_ddi  = in_sizes[15] / 2;
    const int* e_ppi  = (const int*)d_in[16];  int E_ppi  = in_sizes[16] / 2;
    const int* e_mfs  = (const int*)d_in[17];  int E_mfs  = in_sizes[17] / 2;
    const int* e_bps  = (const int*)d_in[18];  int E_bps  = in_sizes[18] / 2;
    const int* e_ccs  = (const int*)d_in[19];  int E_ccs  = in_sizes[19] / 2;
    const int* e_mf2p = (const int*)d_in[20];  int E_mf2p = in_sizes[20] / 2;
    const int* e_bp2p = (const int*)d_in[21];  int E_bp2p = in_sizes[21] / 2;
    const int* e_cc2p = (const int*)d_in[22];  int E_cc2p = in_sizes[22] / 2;
    const int* e_pt2p = (const int*)d_in[23];  int E_pt2p = in_sizes[23] / 2;
    const float* W_dr_emb = (const float*)d_in[24]; const float* b_dr_emb = (const float*)d_in[25];
    const float* W_p_emb  = (const float*)d_in[26]; const float* b_p_emb  = (const float*)d_in[27];
    const float* W_d_emb  = (const float*)d_in[28]; const float* b_d_emb  = (const float*)d_in[29];
    const float* W_mf_emb = (const float*)d_in[30]; const float* b_mf_emb = (const float*)d_in[31];
    const float* W_bp_emb = (const float*)d_in[32]; const float* b_bp_emb = (const float*)d_in[33];
    const float* W_cc_emb = (const float*)d_in[34]; const float* b_cc_emb = (const float*)d_in[35];
    const float* W_pt_emb = (const float*)d_in[36]; const float* b_pt_emb = (const float*)d_in[37];
    const float* W_gcn = (const float*)d_in[38]; const float* b_gcn = (const float*)d_in[39];
    const float* Wp_sg = (const float*)d_in[40]; const float* bp_sg = (const float*)d_in[41];
    const float* Ws_sg = (const float*)d_in[42]; const float* Wn_sg = (const float*)d_in[43];
    const float* b_sg  = (const float*)d_in[44];
    const float* W1 = (const float*)d_in[45]; const float* b1 = (const float*)d_in[46];
    const float* g1 = (const float*)d_in[47]; const float* be1 = (const float*)d_in[48];
    const float* W2 = (const float*)d_in[49]; const float* b2 = (const float*)d_in[50];
    const float* g2 = (const float*)d_in[51]; const float* be2 = (const float*)d_in[52];
    const float* W3 = (const float*)d_in[53]; const float* b3 = (const float*)d_in[54];
    const float* g3 = (const float*)d_in[55]; const float* be3 = (const float*)d_in[56];
    const float* Wo = (const float*)d_in[57]; const float* bo = (const float*)d_in[58];
    float* out = (float*)d_out;

    // ---- workspace carve (floats) ----
    float* ws = (float*)d_ws; size_t off = 0;
    auto alloc = [&](size_t n) { float* p = ws + off; off += n; return p; };
    float* h_dr0 = alloc((size_t)N_DR * H);
    float* h_p0  = alloc((size_t)N_P * H);
    float* h_d   = alloc((size_t)N_DIS * H);
    float* h_mf  = alloc((size_t)N_MF * H);
    float* h_bp  = alloc((size_t)N_BP * H);
    float* h_cc  = alloc((size_t)N_CC * H);
    float* h_pt  = alloc((size_t)N_PATH * H);
    float* mf2   = alloc((size_t)N_MF * H);
    float* bp2   = alloc((size_t)N_BP * H);
    float* cc2   = alloc((size_t)N_CC * H);
    float* h_pgp = alloc((size_t)N_P * H);
    float* aggb  = alloc((size_t)N_P * H);
    float* pool  = alloc((size_t)N_P * H);
    float* dr1   = alloc((size_t)N_DR * H);
    float* p1    = alloc((size_t)N_P * H);
    float* d1    = alloc((size_t)N_DIS * H);
    float* dr2   = alloc((size_t)N_DR * H);
    float* p2    = alloc((size_t)N_P * H);
    float* drnew = alloc((size_t)N_DR * 3 * H);
    float* pnew  = alloc((size_t)N_P * 4 * H);
    float* h1    = alloc((size_t)NB * 256);
    float* h2    = alloc((size_t)NB * 128);
    float* h3    = alloc((size_t)NB * 64);
    float* stats = alloc(512);   // sums[256] + sqs[256]

    auto fill0 = [&](float* p, long n) {
        fill_kernel<<<(unsigned)((n + 255) / 256), 256, 0, stream>>>(p, 0.f, n);
    };
    auto gemm = [&](const float* A1, const int* r1, int lda1, int K1, const float* Wa,
                    const float* A2, const int* r2, int lda2, int K2, const float* Wb,
                    const float* bias, const float* resid, float* o,
                    int M, int N, int flags) {
        dim3 g((unsigned)((N + 63) / 64), (unsigned)((M + 63) / 64));
        wmma_gemm_kernel<<<g, 256, 0, stream>>>(A1, r1, lda1, K1, Wa, A2, r2, lda2, K2, Wb,
                                                bias, resid, o, M, N, flags);
    };
    auto sc_add = [&](const float* feat, const int* e, int E, float* agg) {
        long tot = (long)E * H;
        scatter_add_kernel<<<(unsigned)((tot + 255) / 256), 256, 0, stream>>>(feat, e, e + E, agg, tot);
    };
    auto sc_max = [&](const float* feat, const int* e, int E, float* agg) {
        long tot = (long)E * H;
        scatter_max_kernel<<<(unsigned)((tot + 255) / 256), 256, 0, stream>>>(feat, e, e + E, agg, tot);
    };

    // ---- 1. input embeddings (Linear+ReLU) ----
    gemm(finger, nullptr, 1024, 1024, W_dr_emb, nullptr, nullptr, 0, 0, nullptr,
         b_dr_emb, nullptr, h_dr0, N_DR, H, FLAG_RELU);
    gemm(seqf, nullptr, 400, 400, W_p_emb, nullptr, nullptr, 0, 0, nullptr,
         b_p_emb, nullptr, h_p0, N_P, H, FLAG_RELU);
    gemm(disease, nullptr, 512, 512, W_d_emb, nullptr, nullptr, 0, 0, nullptr,
         b_d_emb, nullptr, h_d, N_DIS, H, FLAG_RELU);
    // identity features: relu(W + b)
    bias_relu_kernel<<<(N_MF * H + 255) / 256, 256, 0, stream>>>(W_mf_emb, b_mf_emb, h_mf, (long)N_MF * H);
    bias_relu_kernel<<<(N_BP * H + 255) / 256, 256, 0, stream>>>(W_bp_emb, b_bp_emb, h_bp, (long)N_BP * H);
    bias_relu_kernel<<<(N_CC * H + 255) / 256, 256, 0, stream>>>(W_cc_emb, b_cc_emb, h_cc, (long)N_CC * H);
    bias_relu_kernel<<<(N_PATH * H + 255) / 256, 256, 0, stream>>>(W_pt_emb, b_pt_emb, h_pt, (long)N_PATH * H);

    // ---- 2. GO similarity GCNs with residual ----
    auto simgcn = [&](const float* hx, const int* e, int E, int Nn, int wi, float* o) {
        fill0(aggb, (long)Nn * H);
        sc_add(hx, e, E, aggb);
        gemm(aggb, nullptr, H, H, W_gcn + (size_t)wi * H * H, nullptr, nullptr, 0, 0, nullptr,
             b_gcn + (size_t)wi * H, hx, o, Nn, H, FLAG_RELU | FLAG_RESID);
    };
    simgcn(h_mf, e_mfs, E_mfs, N_MF, 0, mf2);
    simgcn(h_bp, e_bps, E_bps, N_BP, 1, bp2);
    simgcn(h_cc, e_ccs, E_ccs, N_CC, 2, cc2);

    // ---- 3. GO->protein + pathway->protein, summed ----
    fill0(h_pgp, (long)N_P * H);
    auto gcn2p = [&](const float* hx, const int* e, int E, int wi) {
        fill0(aggb, (long)N_P * H);
        sc_add(hx, e, E, aggb);
        gemm(aggb, nullptr, H, H, W_gcn + (size_t)wi * H * H, nullptr, nullptr, 0, 0, nullptr,
             b_gcn + (size_t)wi * H, nullptr, h_pgp, N_P, H, FLAG_RELU | FLAG_ACCUM);
    };
    gcn2p(mf2, e_mf2p, E_mf2p, 3);
    gcn2p(bp2, e_bp2p, E_bp2p, 4);
    gcn2p(cc2, e_cc2p, E_cc2p, 5);
    gcn2p(h_pt, e_pt2p, E_pt2p, 6);

    // ---- 4. hetero SAGE layers ----
    auto sage = [&](const int* e, int E, const float* xs, int Ns, const float* xd, int Nd,
                    int i, float* ob) {
        gemm(xs, nullptr, H, H, Wp_sg + (size_t)i * H * H, nullptr, nullptr, 0, 0, nullptr,
             bp_sg + (size_t)i * H, nullptr, pool, Ns, H, FLAG_RELU);
        fill0(aggb, (long)Nd * H);
        sc_max(pool, e, E, aggb);
        gemm(xd, nullptr, H, H, Ws_sg + (size_t)i * H * H,
             aggb, nullptr, H, H, Wn_sg + (size_t)i * H * H,
             b_sg + (size_t)i * H, nullptr, ob, Nd, H, FLAG_RELU | FLAG_ACCUM);
    };
    // round 1
    fill0(dr1, (long)N_DR * H); fill0(p1, (long)N_P * H); fill0(d1, (long)N_DIS * H);
    sage(e_dtdr, E_dtdr, h_d, N_DIS, h_dr0, N_DR, 0, dr1);
    sage(e_dmdr, E_dmdr, h_d, N_DIS, h_dr0, N_DR, 1, dr1);
    sage(e_ddi,  E_ddi,  h_dr0, N_DR, h_dr0, N_DR, 6, dr1);
    sage(e_dp,   E_dp,   h_d, N_DIS, h_p0, N_P, 2, p1);
    sage(e_ppi,  E_ppi,  h_p0, N_P, h_p0, N_P, 7, p1);
    sage(e_drtd, E_drtd, h_dr0, N_DR, h_d, N_DIS, 3, d1);
    sage(e_drmd, E_drmd, h_dr0, N_DR, h_d, N_DIS, 4, d1);
    sage(e_pd,   E_pd,   h_p0, N_P, h_d, N_DIS, 5, d1);
    // round 2 (d output unused by the reference -> skipped)
    fill0(dr2, (long)N_DR * H); fill0(p2, (long)N_P * H);
    sage(e_dtdr, E_dtdr, d1, N_DIS, dr1, N_DR, 0, dr2);
    sage(e_dmdr, E_dmdr, d1, N_DIS, dr1, N_DR, 1, dr2);
    sage(e_ddi,  E_ddi,  dr1, N_DR, dr1, N_DR, 6, dr2);
    sage(e_dp,   E_dp,   d1, N_DIS, p1, N_P, 2, p2);
    sage(e_ppi,  E_ppi,  p1, N_P, p1, N_P, 7, p2);

    // ---- 5. concat node features ----
    concat_kernel<<<(unsigned)(((long)N_DR * 3 * H + 255) / 256), 256, 0, stream>>>(
        h_dr0, dr1, dr2, nullptr, drnew, N_DR, 3);
    concat_kernel<<<(unsigned)(((long)N_P * 4 * H + 255) / 256), 256, 0, stream>>>(
        h_p0, p1, p2, h_pgp, pnew, N_P, 4);

    // ---- 6. MLP head with fused row-gather ----
    auto bn = [&](float* x, int C, const float* g, const float* be) {
        fill0(stats, 512);
        bn_stats_kernel<<<NB / 256, 256, 0, stream>>>(x, stats, stats + 256, 256, C);
        long n = (long)NB * C;
        bn_apply_kernel<<<(unsigned)((n + 255) / 256), 256, 0, stream>>>(
            x, stats, stats + 256, g, be, n, C, 1.f / (float)NB);
    };
    gemm(drnew, x_dr, 3 * H, 3 * H, W1,
         pnew, x_p, 4 * H, 4 * H, W1 + (size_t)(3 * H) * 256,
         b1, nullptr, h1, NB, 256, 0);
    bn(h1, 256, g1, be1);
    gemm(h1, nullptr, 256, 256, W2, nullptr, nullptr, 0, 0, nullptr,
         b2, nullptr, h2, NB, 128, 0);
    bn(h2, 128, g2, be2);
    gemm(h2, nullptr, 128, 128, W3, nullptr, nullptr, 0, 0, nullptr,
         b3, nullptr, h3, NB, 64, 0);
    bn(h3, 64, g3, be3);
    out_kernel<<<NB / 256, 256, 0, stream>>>(h3, Wo, bo, out, NB);
}
